// EncoderBaseBlock_42554535968890
// MI455X (gfx1250) — matrix operand order repaired and checked by reading, not compiled, not run on hardware
//
#include <hip/hip_runtime.h>
#include <hip/hip_bf16.h>
#include <stdint.h>

// ---------------------------------------------------------------------------
// EncoderBaseBlock for MI455X (gfx1250, wave32, WMMA).
// bf16 activations + f32 accumulate through v_wmma_f32_16x16x32_bf16.
// ---------------------------------------------------------------------------

#define DEVI __device__ __forceinline__

typedef __attribute__((ext_vector_type(16))) __bf16 v16bf;
typedef __attribute__((ext_vector_type(8)))  float  v8f;

struct U16x16 { uint4 lo, hi; };

DEVI v16bf make_frag(uint4 lo, uint4 hi) {
  U16x16 u{lo, hi};
  return __builtin_bit_cast(v16bf, u);
}

DEVI v8f wmma_bf(v16bf a, v16bf b, v8f c) {
  // 8 args: (neg_a, A, neg_b, B, c_mod, C, reuse_a, reuse_b)
  return __builtin_amdgcn_wmma_f32_16x16x32_bf16(false, a, false, b, (short)0, c,
                                                 false, false);
}

DEVI unsigned f2bf1(float f) {  // round-to-nearest-even f32 -> bf16 bits
  unsigned u = __float_as_uint(f);
  return (u + 0x7FFFu + ((u >> 16) & 1u)) >> 16;
}
DEVI float bf2f(unsigned short h) { return __uint_as_float(((unsigned)h) << 16); }

DEVI uint4 pack8(v8f a) {
  uint4 r;
  r.x = f2bf1(a[0]) | (f2bf1(a[1]) << 16);
  r.y = f2bf1(a[2]) | (f2bf1(a[3]) << 16);
  r.z = f2bf1(a[4]) | (f2bf1(a[5]) << 16);
  r.w = f2bf1(a[6]) | (f2bf1(a[7]) << 16);
  return r;
}

// Problem constants
constexpr int Bn = 8, Cc = 96, Hh = 192, Ww = 192;
constexpr int Nn = Hh * Ww;           // 36864 pixels per image
constexpr int C3 = 288, HEADS = 3, CH = 32, FFNc = 192;
constexpr int TILES_PER_IMG = Nn / 64;  // 576 (64-pixel strips)

// Workspace layout: bf16 (ushort) region then f32 region.
constexpr size_t O_WQKV = 0;                      // 288*96
constexpr size_t O_WPROJ = 27648;                 // 96*96
constexpr size_t O_WC1 = 36864;                   // 192*96
constexpr size_t O_WC2 = 55296;                   // 96*96
constexpr size_t O_ATTN = 64512;                  // 8*3*32*32
constexpr size_t O_QKV = 89088;                   // B*288*N
constexpr size_t QKV_ELEMS = (size_t)Bn * C3 * Nn;
constexpr size_t O_QKVD = O_QKV + QKV_ELEMS;
constexpr size_t BF_TOTAL = O_QKVD + QKV_ELEMS;   // ushorts
constexpr size_t F_Y = 0;                                      // B*96*N floats
constexpr size_t F_G = (size_t)Bn * Cc * Nn;                   // B*3*32*32
constexpr size_t F_SSQ = F_G + (size_t)Bn * HEADS * CH * CH;   // B*96
constexpr size_t F_SSK = F_SSQ + (size_t)Bn * Cc;              // B*96

// ---------------------------------------------------------------------------
// Kernel 0: convert weights to bf16, zero Gram / sum-of-squares accumulators.
// ---------------------------------------------------------------------------
__global__ __launch_bounds__(256) void prep_kernel(
    const float* qkv_w, const float* proj_w, const float* c1_w, const float* c2_w,
    unsigned short* bf, float* G, float* ssq, float* ssk) {
  int i = blockIdx.x * blockDim.x + threadIdx.x;
  int stride = gridDim.x * blockDim.x;
  for (int t = i; t < 288 * 96; t += stride) bf[O_WQKV + t] = (unsigned short)f2bf1(qkv_w[t]);
  for (int t = i; t < 96 * 96; t += stride) bf[O_WPROJ + t] = (unsigned short)f2bf1(proj_w[t]);
  for (int t = i; t < 192 * 96; t += stride) bf[O_WC1 + t] = (unsigned short)f2bf1(c1_w[t]);
  for (int t = i; t < 96 * 96; t += stride) bf[O_WC2 + t] = (unsigned short)f2bf1(c2_w[t]);
  for (int t = i; t < Bn * HEADS * CH * CH; t += stride) G[t] = 0.f;
  for (int t = i; t < Bn * Cc; t += stride) { ssq[t] = 0.f; ssk[t] = 0.f; }
}

// ---------------------------------------------------------------------------
// Kernel 1: fused LayerNorm1 + qkv 1x1-conv GEMM (288x96 @ 96xN) via WMMA.
// Block = 64 pixels x all 288 output channels, 8 waves.
// ---------------------------------------------------------------------------
__global__ __launch_bounds__(256) void ln_qkv_kernel(
    const float* __restrict__ x, const float* ln_w, const float* ln_b,
    const unsigned short* __restrict__ wbf, const float* qkv_b,
    unsigned short* __restrict__ qkv_out) {
  __shared__ unsigned short xnT[64 * 104];  // [pixel][channel] bf16, pitch 104
  __shared__ float redS[4 * 64], redQ[4 * 64], meanS[64], rstdS[64];

  int t = threadIdx.x;
  int b = blockIdx.x / TILES_PER_IMG;
  int n0 = (blockIdx.x % TILES_PER_IMG) * 64;
  int nl = t & 63, cg = t >> 6;

  const float* xb = x + (size_t)b * Cc * Nn + n0;
  float val[24];
  float s = 0.f, q = 0.f;
#pragma unroll
  for (int i = 0; i < 24; i++) {
    int c = cg * 24 + i;
    float v = xb[(size_t)c * Nn + nl];
    val[i] = v; s += v; q += v * v;
  }
  redS[cg * 64 + nl] = s; redQ[cg * 64 + nl] = q;
  __syncthreads();
  if (t < 64) {
    float ts = 0.f, tq = 0.f;
    for (int g = 0; g < 4; g++) { ts += redS[g * 64 + t]; tq += redQ[g * 64 + t]; }
    float mu = ts / Cc;
    meanS[t] = mu;
    rstdS[t] = rsqrtf(tq / Cc - mu * mu + 1e-6f);
  }
  __syncthreads();
  float mu = meanS[nl], rs = rstdS[nl];
#pragma unroll
  for (int i = 0; i < 24; i++) {
    int c = cg * 24 + i;
    float yv = (val[i] - mu) * rs * ln_w[c] + ln_b[c];
    xnT[nl * 104 + c] = (unsigned short)f2bf1(yv);
  }
  __syncthreads();

  int lane = t & 31, wave = t >> 5;
  int l15 = lane & 15, hf = lane >> 4;
  // 18 M-tiles (288/16) x 4 N-tiles = 72 wave tiles
  for (int tile = wave; tile < 72; tile += 8) {
    int mt = tile / 4, nt = tile % 4;
    const unsigned short* arow0 = wbf + (size_t)(mt * 16 + l15) * 96;
    __builtin_prefetch(arow0, 0, 3);  // global_prefetch_b8 on weight row
    v8f acc = {};
#pragma unroll
    for (int kk = 0; kk < 96; kk += 32) {
      const unsigned short* ap = arow0 + kk;
      uint4 a0 = *(const uint4*)(ap + hf * 8);
      uint4 a1 = *(const uint4*)(ap + 16 + hf * 8);
      const unsigned short* bp = &xnT[(nt * 16 + l15) * 104 + kk + hf * 8];
      uint4 b0 = *(const uint4*)bp;
      uint4 b1 = *(const uint4*)(bp + 16);
      acc = wmma_bf(make_frag(a0, a1), make_frag(b0, b1), acc);
    }
#pragma unroll
    for (int r = 0; r < 8; r++) {
      int row = mt * 16 + hf * 8 + r;
      int col = n0 + nt * 16 + l15;
      float v = acc[r] + qkv_b[row];
      qkv_out[((size_t)b * C3 + row) * Nn + col] = (unsigned short)f2bf1(v);
    }
  }
}

// ---------------------------------------------------------------------------
// Kernel 2: depthwise 3x3 (SAME, zero pad) + partial Gram q.k^T via WMMA
// (reduction dim = pixels) + row sum-of-squares, atomically accumulated.
// ---------------------------------------------------------------------------
__global__ __launch_bounds__(256) void dw_gram_kernel(
    const unsigned short* __restrict__ qkv, const float* dw_w, const float* dw_b,
    unsigned short* __restrict__ qkvd, float* G, float* ssq, float* ssk) {
  __shared__ unsigned short sh[288 * 64];  // [channel][pixel] bf16
  int t = threadIdx.x;
  int b = blockIdx.x / TILES_PER_IMG;
  int ti = blockIdx.x % TILES_PER_IMG;
  int hh = ti / 3, w0 = (ti % 3) * 64;
  int nl = t & 63, cg = t >> 6;
  int w = w0 + nl;

  const unsigned short* qb = qkv + (size_t)b * C3 * Nn;
  for (int c = cg; c < C3; c += 4) {
    const unsigned short* qc = qb + (size_t)c * Nn;
    const float* wc = dw_w + c * 9;
    float acc = dw_b[c];
#pragma unroll
    for (int dy = -1; dy <= 1; dy++) {
      int hy = hh + dy;
      if ((unsigned)hy < (unsigned)Hh) {
#pragma unroll
        for (int dx = -1; dx <= 1; dx++) {
          int wx = w + dx;
          if ((unsigned)wx < (unsigned)Ww)
            acc += bf2f(qc[hy * Ww + wx]) * wc[(dy + 1) * 3 + (dx + 1)];
        }
      }
    }
    unsigned short hv = (unsigned short)f2bf1(acc);
    sh[c * 64 + nl] = hv;
    qkvd[((size_t)b * C3 + c) * Nn + hh * Ww + w] = hv;
  }
  __syncthreads();

  // Row sum-of-squares for q (channels 0..95) and k (96..191).
  if (t < 192) {
    float s = 0.f;
    for (int n = 0; n < 64; n++) { float v = bf2f(sh[t * 64 + n]); s += v * v; }
    if (t < 96) atomicAdd(&ssq[b * 96 + t], s);
    else        atomicAdd(&ssk[b * 96 + (t - 96)], s);
  }

  // Partial Gram: G[b,h,c,d] += sum_n q[c,n] * k[d,n], K = 64 pixels.
  int lane = t & 31, wave = t >> 5;
  int l15 = lane & 15, hf = lane >> 4;
  for (int tile = wave; tile < 12; tile += 8) {
    int head = tile >> 2, ct = (tile >> 1) & 1, dt = tile & 1;
    v8f acc = {};
#pragma unroll
    for (int kk = 0; kk < 64; kk += 32) {
      const unsigned short* ap = &sh[(head * 32 + ct * 16 + l15) * 64 + kk];
      uint4 a0 = *(const uint4*)(ap + hf * 8);
      uint4 a1 = *(const uint4*)(ap + 16 + hf * 8);
      const unsigned short* bp =
          &sh[(96 + head * 32 + dt * 16 + l15) * 64 + kk + hf * 8];
      uint4 b0 = *(const uint4*)bp;
      uint4 b1 = *(const uint4*)(bp + 16);
      acc = wmma_bf(make_frag(a0, a1), make_frag(b0, b1), acc);
    }
#pragma unroll
    for (int r = 0; r < 8; r++) {
      int crow = ct * 16 + hf * 8 + r, dcol = dt * 16 + l15;
      atomicAdd(&G[(((size_t)b * HEADS + head) * CH + crow) * CH + dcol], acc[r]);
    }
  }
}

// ---------------------------------------------------------------------------
// Kernel 3: finalize attention: l2-normalize scale, *temp, relu, softmax.
// One wave per (batch, head); thread = query channel c.
// ---------------------------------------------------------------------------
__global__ __launch_bounds__(32) void attn_kernel(
    const float* G, const float* ssq, const float* ssk, const float* temp,
    unsigned short* attn) {
  int b = blockIdx.x / HEADS, h = blockIdx.x % HEADS;
  int c = threadIdx.x;  // 0..31
  const float* g = G + (((size_t)b * HEADS + h) * CH + c) * CH;
  float rq = fmaxf(sqrtf(ssq[b * 96 + h * CH + c]), 1e-12f);
  float tmp = temp[h];
  float a[32];
  float m = -1e30f;
  for (int d = 0; d < CH; d++) {
    float rk = fmaxf(sqrtf(ssk[b * 96 + h * CH + d]), 1e-12f);
    float v = (g[d] / (rq * rk)) * tmp;
    v = fmaxf(v, 0.f);
    a[d] = v; m = fmaxf(m, v);
  }
  float s = 0.f;
  for (int d = 0; d < CH; d++) { a[d] = __expf(a[d] - m); s += a[d]; }
  float inv = 1.f / s;
  unsigned short* ar = attn + (((size_t)b * HEADS + h) * CH + c) * CH;
  for (int d = 0; d < CH; d++) ar[d] = (unsigned short)f2bf1(a[d] * inv);
}

// ---------------------------------------------------------------------------
// Kernel 4: out1 = A @ v per head (WMMA, K=32), then proj 96x96 WMMA,
// then residual y = x + betac * xc.
// ---------------------------------------------------------------------------
__global__ __launch_bounds__(256) void av_proj_kernel(
    const unsigned short* __restrict__ qkvd, const unsigned short* __restrict__ attn,
    const unsigned short* __restrict__ wproj, const float* proj_b,
    const float* betac, const float* __restrict__ x, float* __restrict__ y) {
  __shared__ unsigned short vT[64 * 104];   // [pixel][v-channel]
  __shared__ unsigned short o1T[64 * 104];  // [pixel][attn-out channel]
  int t = threadIdx.x;
  int b = blockIdx.x / TILES_PER_IMG;
  int n0 = (blockIdx.x % TILES_PER_IMG) * 64;
  int nl = t & 63, cg = t >> 6;

  const unsigned short* vb = qkvd + ((size_t)b * C3 + 192) * Nn + n0;
#pragma unroll
  for (int i = 0; i < 24; i++) {
    int c = cg * 24 + i;
    vT[nl * 104 + c] = vb[(size_t)c * Nn + nl];
  }
  __syncthreads();

  int lane = t & 31, wave = t >> 5;
  int l15 = lane & 15, hf = lane >> 4;

  // out1: 3 heads x 2 M-tiles x 4 N-tiles = 24 tiles, 1 WMMA each (K=32)
  for (int tile = wave; tile < 24; tile += 8) {
    int head = tile / 8, rem = tile % 8, mt = rem / 4, nt = rem % 4;
    const unsigned short* ap =
        attn + (((size_t)b * HEADS + head) * CH + mt * 16 + l15) * CH;
    uint4 a0 = *(const uint4*)(ap + hf * 8);
    uint4 a1 = *(const uint4*)(ap + 16 + hf * 8);
    const unsigned short* bp = &vT[(nt * 16 + l15) * 104 + head * 32 + hf * 8];
    uint4 b0 = *(const uint4*)bp;
    uint4 b1 = *(const uint4*)(bp + 16);
    v8f acc = {};
    acc = wmma_bf(make_frag(a0, a1), make_frag(b0, b1), acc);
    *(uint4*)&o1T[(nt * 16 + l15) * 104 + head * 32 + mt * 16 + hf * 8] = pack8(acc);
  }
  __syncthreads();

  // proj: 6 M-tiles x 4 N-tiles, K = 96
  for (int tile = wave; tile < 24; tile += 8) {
    int mt = tile / 4, nt = tile % 4;
    const unsigned short* arow0 = wproj + (size_t)(mt * 16 + l15) * 96;
    __builtin_prefetch(arow0, 0, 3);
    v8f acc = {};
#pragma unroll
    for (int kk = 0; kk < 96; kk += 32) {
      const unsigned short* ap = arow0 + kk;
      uint4 a0 = *(const uint4*)(ap + hf * 8);
      uint4 a1 = *(const uint4*)(ap + 16 + hf * 8);
      const unsigned short* bp = &o1T[(nt * 16 + l15) * 104 + kk + hf * 8];
      uint4 b0 = *(const uint4*)bp;
      uint4 b1 = *(const uint4*)(bp + 16);
      acc = wmma_bf(make_frag(a0, a1), make_frag(b0, b1), acc);
    }
#pragma unroll
    for (int r = 0; r < 8; r++) {
      int row = mt * 16 + hf * 8 + r;
      int col = n0 + nt * 16 + l15;
      size_t idx = ((size_t)b * Cc + row) * Nn + col;
      y[idx] = x[idx] + betac[row] * (acc[r] + proj_b[row]);
    }
  }
}

// ---------------------------------------------------------------------------
// Kernel 5: fused LayerNorm2 + conv1 (192x96) + gate f1*f2 + conv2 (96x96)
// + final residual with gamma.
// ---------------------------------------------------------------------------
__global__ __launch_bounds__(256) void ffn_kernel(
    const float* __restrict__ y, const float* ln_w, const float* ln_b,
    const unsigned short* __restrict__ wc1, const float* c1_b,
    const unsigned short* __restrict__ wc2, const float* c2_b,
    const float* gamma, float* __restrict__ out) {
  __shared__ unsigned short ynT[64 * 104];  // normalized y, [pixel][c]
  __shared__ unsigned short fT[64 * 200];   // conv1 output, [pixel][0..191]
  __shared__ unsigned short gT[64 * 104];   // gated, [pixel][c]
  __shared__ float redS[4 * 64], redQ[4 * 64], meanS[64], rstdS[64];

  int t = threadIdx.x;
  int b = blockIdx.x / TILES_PER_IMG;
  int n0 = (blockIdx.x % TILES_PER_IMG) * 64;
  int nl = t & 63, cg = t >> 6;

  const float* yb = y + (size_t)b * Cc * Nn + n0;
  float val[24];
  float s = 0.f, q = 0.f;
#pragma unroll
  for (int i = 0; i < 24; i++) {
    int c = cg * 24 + i;
    float v = yb[(size_t)c * Nn + nl];
    val[i] = v; s += v; q += v * v;
  }
  redS[cg * 64 + nl] = s; redQ[cg * 64 + nl] = q;
  __syncthreads();
  if (t < 64) {
    float ts = 0.f, tq = 0.f;
    for (int g = 0; g < 4; g++) { ts += redS[g * 64 + t]; tq += redQ[g * 64 + t]; }
    float mu = ts / Cc;
    meanS[t] = mu;
    rstdS[t] = rsqrtf(tq / Cc - mu * mu + 1e-6f);
  }
  __syncthreads();
  float mu = meanS[nl], rs = rstdS[nl];
#pragma unroll
  for (int i = 0; i < 24; i++) {
    int c = cg * 24 + i;
    float yv = (val[i] - mu) * rs * ln_w[c] + ln_b[c];
    ynT[nl * 104 + c] = (unsigned short)f2bf1(yv);
  }
  __syncthreads();

  int lane = t & 31, wave = t >> 5;
  int l15 = lane & 15, hf = lane >> 4;

  // conv1: 12 M-tiles (192/16) x 4 N-tiles = 48 tiles, K = 96
  for (int tile = wave; tile < 48; tile += 8) {
    int mt = tile / 4, nt = tile % 4;
    const unsigned short* arow0 = wc1 + (size_t)(mt * 16 + l15) * 96;
    __builtin_prefetch(arow0, 0, 3);
    v8f acc = {};
#pragma unroll
    for (int kk = 0; kk < 96; kk += 32) {
      const unsigned short* ap = arow0 + kk;
      uint4 a0 = *(const uint4*)(ap + hf * 8);
      uint4 a1 = *(const uint4*)(ap + 16 + hf * 8);
      const unsigned short* bp = &ynT[(nt * 16 + l15) * 104 + kk + hf * 8];
      uint4 b0 = *(const uint4*)bp;
      uint4 b1 = *(const uint4*)(bp + 16);
      acc = wmma_bf(make_frag(a0, a1), make_frag(b0, b1), acc);
    }
    v8f fv;
#pragma unroll
    for (int r = 0; r < 8; r++) fv[r] = acc[r] + c1_b[mt * 16 + hf * 8 + r];
    *(uint4*)&fT[(nt * 16 + l15) * 200 + mt * 16 + hf * 8] = pack8(fv);
  }
  __syncthreads();

  // gate: g = f1 * f2
#pragma unroll
  for (int i = 0; i < 24; i++) {
    int c = cg * 24 + i;
    float g = bf2f(fT[nl * 200 + c]) * bf2f(fT[nl * 200 + 96 + c]);
    gT[nl * 104 + c] = (unsigned short)f2bf1(g);
  }
  __syncthreads();

  // conv2: 6 M-tiles x 4 N-tiles, K = 96; epilogue = y + gamma * f
  for (int tile = wave; tile < 24; tile += 8) {
    int mt = tile / 4, nt = tile % 4;
    const unsigned short* arow0 = wc2 + (size_t)(mt * 16 + l15) * 96;
    v8f acc = {};
#pragma unroll
    for (int kk = 0; kk < 96; kk += 32) {
      const unsigned short* ap = arow0 + kk;
      uint4 a0 = *(const uint4*)(ap + hf * 8);
      uint4 a1 = *(const uint4*)(ap + 16 + hf * 8);
      const unsigned short* bp = &gT[(nt * 16 + l15) * 104 + kk + hf * 8];
      uint4 b0 = *(const uint4*)bp;
      uint4 b1 = *(const uint4*)(bp + 16);
      acc = wmma_bf(make_frag(a0, a1), make_frag(b0, b1), acc);
    }
#pragma unroll
    for (int r = 0; r < 8; r++) {
      int row = mt * 16 + hf * 8 + r;
      int col = n0 + nt * 16 + l15;
      size_t idx = ((size_t)b * Cc + row) * Nn + col;
      out[idx] = y[idx] + gamma[row] * (acc[r] + c2_b[row]);
    }
  }
}

// ---------------------------------------------------------------------------
extern "C" void kernel_launch(void* const* d_in, const int* in_sizes, int n_in,
                              void* d_out, int out_size, void* d_ws, size_t ws_size,
                              hipStream_t stream) {
  (void)in_sizes; (void)n_in; (void)out_size; (void)ws_size;
  const float* x      = (const float*)d_in[0];
  const float* ln1_w  = (const float*)d_in[1];
  const float* ln1_b  = (const float*)d_in[2];
  const float* ln2_w  = (const float*)d_in[3];
  const float* ln2_b  = (const float*)d_in[4];
  const float* qkv_w  = (const float*)d_in[5];
  const float* qkv_b  = (const float*)d_in[6];
  const float* dw_w   = (const float*)d_in[7];
  const float* dw_b   = (const float*)d_in[8];
  const float* temp   = (const float*)d_in[9];
  const float* proj_w = (const float*)d_in[10];
  const float* proj_b = (const float*)d_in[11];
  const float* c1_w   = (const float*)d_in[12];
  const float* c1_b   = (const float*)d_in[13];
  const float* c2_w   = (const float*)d_in[14];
  const float* c2_b   = (const float*)d_in[15];
  const float* betac  = (const float*)d_in[16];
  const float* gamma  = (const float*)d_in[17];

  unsigned short* bf = (unsigned short*)d_ws;
  float* fregion = (float*)((char*)d_ws + BF_TOTAL * sizeof(unsigned short));
  float* Y   = fregion + F_Y;
  float* G   = fregion + F_G;
  float* SSQ = fregion + F_SSQ;
  float* SSK = fregion + F_SSK;

  int nblk = Bn * TILES_PER_IMG;  // 4608
  prep_kernel<<<256, 256, 0, stream>>>(qkv_w, proj_w, c1_w, c2_w, bf, G, SSQ, SSK);
  ln_qkv_kernel<<<nblk, 256, 0, stream>>>(x, ln1_w, ln1_b, bf + O_WQKV, qkv_b,
                                          bf + O_QKV);
  dw_gram_kernel<<<nblk, 256, 0, stream>>>(bf + O_QKV, dw_w, dw_b, bf + O_QKVD,
                                           G, SSQ, SSK);
  attn_kernel<<<Bn * HEADS, 32, 0, stream>>>(G, SSQ, SSK, temp, bf + O_ATTN);
  av_proj_kernel<<<nblk, 256, 0, stream>>>(bf + O_QKVD, bf + O_ATTN, bf + O_WPROJ,
                                           proj_b, betac, x, Y);
  ffn_kernel<<<nblk, 256, 0, stream>>>(Y, ln2_w, ln2_b, bf + O_WC1, c1_b,
                                       bf + O_WC2, c2_b, gamma, (float*)d_out);
}
